// FastformerAttention_13889924235607
// MI455X (gfx1250) — matrix operand order repaired and checked by reading, compiled, not run on hardware
//
#include <hip/hip_runtime.h>
#include <hip/hip_bf16.h>
#include <stdint.h>

// ---------------- problem constants ----------------
#define BATCH     16
#define SEQ       2048
#define EMB       1024
#define NHEADS    16
#define HEAD_D    64          // EMB / NHEADS
#define MROWS     (BATCH * SEQ)   // 32768
#define NEG_FILL  (-10000.0f)
#define INV_SCALE 0.125f      // 1/sqrt(64)

// ---------------- WMMA vector types ----------------
typedef __attribute__((ext_vector_type(16))) __bf16 v16bf;
typedef __attribute__((ext_vector_type(8)))  __bf16 v8bf;
typedef __attribute__((ext_vector_type(8)))  float  v8f;

// ---------------- helpers ----------------
__device__ __forceinline__ unsigned short f32_to_bf16_raw(float f) {
    unsigned int u = __float_as_uint(f);
    unsigned int r = u + 0x7FFFu + ((u >> 16) & 1u);   // round-to-nearest-even
    return (unsigned short)(r >> 16);
}

// fp32 -> raw bf16, 4 elements/thread (n must be divisible by 4)
__global__ __launch_bounds__(256) void cvt_f32_to_bf16(const float* __restrict__ in,
                                                       unsigned short* __restrict__ out,
                                                       long long n) {
    long long i = ((long long)blockIdx.x * blockDim.x + threadIdx.x) * 4;
    if (i < n) {
        float4 f = *(const float4*)(in + i);
        ushort4 o;
        o.x = f32_to_bf16_raw(f.x);
        o.y = f32_to_bf16_raw(f.y);
        o.z = f32_to_bf16_raw(f.z);
        o.w = f32_to_bf16_raw(f.w);
        *(ushort4*)(out + i) = o;
    }
}

// ---------------- WMMA GEMM ----------------
// C[M,N] = A[M,K](bf16) * W[N,K](bf16)^T + bias[N], C fp32.
// Block tile 128x256, 8 waves, wave tile 64x64 (4x4 WMMA 16x16 tiles):
// 16 v_wmma per K-step against 16 b128 loads (4-way A and B reuse).
__global__ __launch_bounds__(256) void gemm_bf16_wmma(const unsigned short* __restrict__ A,
                                                      const unsigned short* __restrict__ W,
                                                      const float* __restrict__ bias,
                                                      float* __restrict__ C,
                                                      int M, int N, int K) {
    const int wave = threadIdx.x >> 5;          // wave32
    const int lane = threadIdx.x & 31;
    const int wm   = wave >> 2;                 // 0..1 -> 64-row group
    const int wn   = wave & 3;                  // 0..3 -> 64-col group
    const int m0   = blockIdx.y * 128 + wm * 64;
    const int n0   = blockIdx.x * 256 + wn * 64;
    const int r    = lane & 15;                 // row/col within 16
    const int g    = lane >> 4;                 // lane half

    const __bf16* Ab = (const __bf16*)A;
    const __bf16* Wb = (const __bf16*)W;

    // per-lane base pointers (K-advance is the only loop-carried address math)
    const __bf16* ap[4];
    const __bf16* bp[4];
#pragma unroll
    for (int i = 0; i < 4; ++i) ap[i] = Ab + (size_t)(m0 + i * 16 + r) * K + g * 8;
#pragma unroll
    for (int j = 0; j < 4; ++j) bp[j] = Wb + (size_t)(n0 + j * 16 + r) * K + g * 8;

    v8f acc[4][4];
#pragma unroll
    for (int i = 0; i < 4; ++i)
#pragma unroll
        for (int j = 0; j < 4; ++j) acc[i][j] = (v8f){0.f,0.f,0.f,0.f,0.f,0.f,0.f,0.f};

    for (int k0 = 0; k0 < K; k0 += 32) {
        // speculative prefetch of next K tile (silently dropped past the end)
        __builtin_prefetch(ap[0] + k0 + 32, 0, 1);
        __builtin_prefetch(bp[0] + k0 + 32, 0, 1);

        // A fragments: 16x32 bf16. lane half g holds K = g*8..g*8+7 (VGPR0-3)
        // and K = 16+g*8..+7 (VGPR4-7), row M = lane&15.   [ISA 7.12.2]
        v16bf a[4];
#pragma unroll
        for (int i = 0; i < 4; ++i) {
            const __bf16* p = ap[i] + k0;
            v8bf lo = *(const v8bf*)(p);        // K = g*8 .. +7
            v8bf hi = *(const v8bf*)(p + 16);   // K = 16+g*8 .. +7
            a[i] = __builtin_shufflevector(lo, hi, 0,1,2,3,4,5,6,7,8,9,10,11,12,13,14,15);
        }
        // B fragments: 32x16 bf16, column n = W row (B = W^T). lane half g
        // holds K = g*8..g*8+7 and K = 16+g*8..+7 of W row (n0+j*16+r).
        v16bf bm[4];
#pragma unroll
        for (int j = 0; j < 4; ++j) {
            const __bf16* p = bp[j] + k0;
            v8bf lo = *(const v8bf*)(p);
            v8bf hi = *(const v8bf*)(p + 16);
            bm[j] = __builtin_shufflevector(lo, hi, 0,1,2,3,4,5,6,7,8,9,10,11,12,13,14,15);
        }
#pragma unroll
        for (int i = 0; i < 4; ++i)
#pragma unroll
            for (int j = 0; j < 4; ++j)
                acc[i][j] = __builtin_amdgcn_wmma_f32_16x16x32_bf16(
                    false, a[i], false, bm[j], (short)0, acc[i][j], false, false);
    }

    // C/D layout: VGPR vi holds row g*8+vi of the 16x16 tile, col = lane&15.
#pragma unroll
    for (int i = 0; i < 4; ++i) {
#pragma unroll
        for (int j = 0; j < 4; ++j) {
            const int col  = n0 + j * 16 + r;
            const float bc = bias[col];
#pragma unroll
            for (int vi = 0; vi < 8; ++vi) {
                const int row = m0 + i * 16 + g * 8 + vi;
                C[(size_t)row * N + col] = acc[i][j][vi] + bc;
            }
        }
    }
}

// ---------------- attention pooling ----------------
__device__ __forceinline__ float block_max(float v, float* red) {
    int t = threadIdx.x;
    red[t] = v; __syncthreads();
    for (int o = 128; o > 0; o >>= 1) {
        if (t < o) red[t] = fmaxf(red[t], red[t + o]);
        __syncthreads();
    }
    float r = red[0]; __syncthreads();
    return r;
}
__device__ __forceinline__ float block_sum(float v, float* red) {
    int t = threadIdx.x;
    red[t] = v; __syncthreads();
    for (int o = 128; o > 0; o >>= 1) {
        if (t < o) red[t] += red[t + o];
        __syncthreads();
    }
    float r = red[0]; __syncthreads();
    return r;
}

// one block per (b,h); q/k/v are (B,S,E) fp32 with head h at cols h*64..
__global__ __launch_bounds__(256) void fastformer_pool(const float* __restrict__ q,
                                                       const float* __restrict__ k,
                                                       const float* __restrict__ v,
                                                       const int* __restrict__ mask,
                                                       unsigned short* __restrict__ attn) {
    __shared__ float s_logit[SEQ];
    __shared__ float s_red[256];
    __shared__ float s_glob[HEAD_D];
    __shared__ float s_part[4 * HEAD_D];

    const int b   = blockIdx.x / NHEADS;
    const int h   = blockIdx.x % NHEADS;
    const int tid = threadIdx.x;
    const float* qb = q + ((size_t)b * SEQ) * EMB + h * HEAD_D;
    const float* kb = k + ((size_t)b * SEQ) * EMB + h * HEAD_D;
    const float* vb = v + ((size_t)b * SEQ) * EMB + h * HEAD_D;
    const int*   mb = mask + b * SEQ;
    const int d   = tid & 63;
    const int grp = tid >> 6;   // 0..3

    // ---- phase 1: q_w = softmax(sum_d q / sqrt(D)); q_global = sum_s q_w * q
    float lmax = -3.0e38f;
    for (int s = tid; s < SEQ; s += 256) {
        const float* row = qb + (size_t)s * EMB;
        float sum = 0.f;
#pragma unroll
        for (int dd = 0; dd < HEAD_D; ++dd) sum += row[dd];
        float lg = mb[s] ? sum * INV_SCALE : NEG_FILL;
        s_logit[s] = lg;
        lmax = fmaxf(lmax, lg);
    }
    float gmax = block_max(lmax, s_red);
    float lsum = 0.f;
    for (int s = tid; s < SEQ; s += 256) {
        float e = __expf(s_logit[s] - gmax);
        s_logit[s] = e;
        lsum += e;
    }
    float Z = block_sum(lsum, s_red);
    {
        float acc = 0.f;
        for (int s = grp; s < SEQ; s += 4) acc += s_logit[s] * qb[(size_t)s * EMB + d];
        s_part[grp * HEAD_D + d] = acc;
    }
    __syncthreads();
    if (tid < HEAD_D)
        s_glob[tid] = (s_part[tid] + s_part[HEAD_D + tid] +
                       s_part[2 * HEAD_D + tid] + s_part[3 * HEAD_D + tid]) / Z;
    __syncthreads();

    // ---- phase 2: k_w = softmax(k . q_global); k_global = sum_s k_w * k
    lmax = -3.0e38f;
    for (int s = tid; s < SEQ; s += 256) {
        const float* row = kb + (size_t)s * EMB;
        float dot = 0.f;
#pragma unroll
        for (int dd = 0; dd < HEAD_D; ++dd) dot += row[dd] * s_glob[dd];
        float lg = mb[s] ? dot : NEG_FILL;
        s_logit[s] = lg;
        lmax = fmaxf(lmax, lg);
    }
    gmax = block_max(lmax, s_red);
    lsum = 0.f;
    for (int s = tid; s < SEQ; s += 256) {
        float e = __expf(s_logit[s] - gmax);
        s_logit[s] = e;
        lsum += e;
    }
    Z = block_sum(lsum, s_red);
    {
        float acc = 0.f;
        for (int s = grp; s < SEQ; s += 4) acc += s_logit[s] * kb[(size_t)s * EMB + d];
        s_part[grp * HEAD_D + d] = acc;
    }
    __syncthreads();
    if (tid < HEAD_D)
        s_glob[tid] = (s_part[tid] + s_part[HEAD_D + tid] +
                       s_part[2 * HEAD_D + tid] + s_part[3 * HEAD_D + tid]) / Z;
    __syncthreads();

    // ---- phase 3: v_w = softmax(q . k_global); attn = v_w * v  (store bf16)
    lmax = -3.0e38f;
    for (int s = tid; s < SEQ; s += 256) {
        const float* row = qb + (size_t)s * EMB;
        float dot = 0.f;
#pragma unroll
        for (int dd = 0; dd < HEAD_D; ++dd) dot += row[dd] * s_glob[dd];
        float lg = mb[s] ? dot : NEG_FILL;
        s_logit[s] = lg;
        lmax = fmaxf(lmax, lg);
    }
    gmax = block_max(lmax, s_red);
    lsum = 0.f;
    for (int s = tid; s < SEQ; s += 256) {
        float e = __expf(s_logit[s] - gmax);
        s_logit[s] = e;
        lsum += e;
    }
    Z = block_sum(lsum, s_red);
    const float invZ = 1.0f / Z;
    for (int s = tid; s < SEQ; s += 256) {
        const float w = s_logit[s] * invZ;
        const float* row = vb + (size_t)s * EMB;
        unsigned short* orow = attn + ((size_t)(b * SEQ + s)) * EMB + h * HEAD_D;
#pragma unroll
        for (int dd = 0; dd < HEAD_D; dd += 4) {
            float4 f = *(const float4*)(row + dd);
            ushort4 o;
            o.x = f32_to_bf16_raw(w * f.x);
            o.y = f32_to_bf16_raw(w * f.y);
            o.z = f32_to_bf16_raw(w * f.z);
            o.w = f32_to_bf16_raw(w * f.w);
            *(ushort4*)(orow + dd) = o;
        }
    }
}

// ---------------- launch ----------------
extern "C" void kernel_launch(void* const* d_in, const int* in_sizes, int n_in,
                              void* d_out, int out_size, void* d_ws, size_t ws_size,
                              hipStream_t stream) {
    (void)in_sizes; (void)n_in; (void)out_size; (void)ws_size;
    const float* x    = (const float*)d_in[0];
    const int*   mask = (const int*)d_in[1];
    const float* Wq   = (const float*)d_in[2];
    const float* bq   = (const float*)d_in[3];
    const float* Wk   = (const float*)d_in[4];
    const float* bk   = (const float*)d_in[5];
    const float* Wv   = (const float*)d_in[6];
    const float* bv   = (const float*)d_in[7];
    const float* Wo   = (const float*)d_in[8];
    const float* bo   = (const float*)d_in[9];
    float* out = (float*)d_out;

    // workspace layout (bytes)
    const size_t n_x  = (size_t)MROWS * EMB;     // 33,554,432
    const size_t n_w  = (size_t)EMB * EMB;       // 1,048,576
    char* ws = (char*)d_ws;
    unsigned short* xb   = (unsigned short*)(ws);
    unsigned short* wqb  = (unsigned short*)(ws + n_x * 2);
    unsigned short* wkb  = (unsigned short*)(ws + n_x * 2 + n_w * 2);
    unsigned short* wvb  = (unsigned short*)(ws + n_x * 2 + n_w * 4);
    unsigned short* wob  = (unsigned short*)(ws + n_x * 2 + n_w * 6);
    float* qf            = (float*)(ws + n_x * 2 + n_w * 8);
    float* kf            = qf + n_x;
    float* vf            = kf + n_x;
    unsigned short* attn = (unsigned short*)(vf + n_x);

    // 1) fp32 -> bf16 conversions
    cvt_f32_to_bf16<<<(unsigned)(n_x / 4 / 256), 256, 0, stream>>>(x,  xb,  (long long)n_x);
    cvt_f32_to_bf16<<<(unsigned)(n_w / 4 / 256), 256, 0, stream>>>(Wq, wqb, (long long)n_w);
    cvt_f32_to_bf16<<<(unsigned)(n_w / 4 / 256), 256, 0, stream>>>(Wk, wkb, (long long)n_w);
    cvt_f32_to_bf16<<<(unsigned)(n_w / 4 / 256), 256, 0, stream>>>(Wv, wvb, (long long)n_w);
    cvt_f32_to_bf16<<<(unsigned)(n_w / 4 / 256), 256, 0, stream>>>(Wo, wob, (long long)n_w);

    // 2) q/k/v projections: (32768,1024) x (1024,1024)^T
    dim3 ggrid(EMB / 256, MROWS / 128);
    gemm_bf16_wmma<<<ggrid, 256, 0, stream>>>(xb, wqb, bq, qf, MROWS, EMB, EMB);
    gemm_bf16_wmma<<<ggrid, 256, 0, stream>>>(xb, wkb, bk, kf, MROWS, EMB, EMB);
    gemm_bf16_wmma<<<ggrid, 256, 0, stream>>>(xb, wvb, bv, vf, MROWS, EMB, EMB);

    // 3) fastformer pooling, one block per (b,h)
    fastformer_pool<<<BATCH * NHEADS, 256, 0, stream>>>(qf, kf, vf, mask, attn);

    // 4) output projection into d_out
    gemm_bf16_wmma<<<ggrid, 256, 0, stream>>>(attn, wob, bo, out, MROWS, EMB, EMB);
}
